// MultiHeadedEMA_5772436046422
// MI455X (gfx1250) — matrix operand order repaired and checked by reading, not compiled, not run on hardware
//
#include <hip/hip_runtime.h>
#include <math.h>

#define HEADS_N 8
#define DIMSZ   1024
#define BSZ     2
#define LSEQ    2048
#define SEG     256            // timesteps per segment
#define NSEG    (LSEQ / SEG)   // 8
#define CH      16             // WMMA chunk (timesteps)
#define NCH     (SEG / CH)     // 16 chunks per segment
#define DT      16             // d-channels per tile (WMMA N)

typedef _Float16 half16 __attribute__((ext_vector_type(16)));
typedef _Float16 half8  __attribute__((ext_vector_type(8)));
typedef float    float8 __attribute__((ext_vector_type(8)));

__device__ __forceinline__ float sigm(float x) { return 1.0f / (1.0f + __expf(-x)); }

__device__ __forceinline__ size_t CIDX(int b, int dir, int s, int h, int d) {
    return ((((size_t)(b * 2 + dir) * NSEG + s) * HEADS_N) + h) * DIMSZ + d;
}

// ---------------------------------------------------------------------------
// K1: per-(b, d-tile, segment) local chunked scan via WMMA.
// wave0 = forward heads 0..7, wave1 = reverse heads 8..15.
// Writes: out (local part, both directions summed), cws (weighted segment carries).
// ---------------------------------------------------------------------------
__global__ __launch_bounds__(64) void k1_local(
    const float* __restrict__ x,  const float* __restrict__ E,
    const float* __restrict__ R,
    const float* __restrict__ al, const float* __restrict__ da,
    const float* __restrict__ ral,const float* __restrict__ rda,
    float* __restrict__ out, float* __restrict__ cws)
{
    const int blk = blockIdx.x;
    const int dt  = blk % (DIMSZ / DT);
    const int s   = (blk / (DIMSZ / DT)) % NSEG;
    const int b   = blk / ((DIMSZ / DT) * NSEG);

    __shared__ __align__(16) _Float16 xs[DT][SEG + 8];   // transposed x tile (f16)
    __shared__ __align__(16) float Obuf[SEG][DT + 1];    // fp32 output tile

    // ---- stage x tile: b128 loads, transpose into f16 LDS ----
    const float* xb = x + ((size_t)b * LSEQ + (size_t)s * SEG) * DIMSZ + (size_t)dt * DT;
    for (int i = threadIdx.x; i < SEG * DT / 4; i += 64) {
        const int t = i >> 2, c4 = (i & 3) * 4;
        const float4 v = *(const float4*)&xb[(size_t)t * DIMSZ + c4];
        xs[c4 + 0][t] = (_Float16)v.x;
        xs[c4 + 1][t] = (_Float16)v.y;
        xs[c4 + 2][t] = (_Float16)v.z;
        xs[c4 + 3][t] = (_Float16)v.w;
    }
    for (int i = threadIdx.x; i < SEG * (DT + 1); i += 64)
        ((float*)Obuf)[i] = 0.0f;
    __syncthreads();

    const int wave = threadIdx.x >> 5;      // 0 fwd, 1 rev
    const int lane = threadIdx.x & 31;
    const int hl   = lane >> 4;             // half-wave (head-group select)
    const int dl   = lane & 15;             // local d column (WMMA N)
    const int row  = lane & 15;             // WMMA A row (M)

    const float* Aal = wave ? ral : al;
    const float* Ada = wave ? rda : da;

    // per-head scan parameters (uniform across lanes -> SGPRs)
    float a_[8], q_[8], l2q_[8], q9_[8], q16_[8];
#pragma unroll
    for (int h = 0; h < HEADS_N; ++h) {
        const float av = sigm(Aal[h]);
        const float qv = (1.0f - av) * sigm(Ada[h]);
        a_[h] = av; q_[h] = qv;
        l2q_[h] = __log2f(qv);
        q9_[h]  = exp2f(9.0f  * l2q_[h]);
        q16_[h] = exp2f(16.0f * l2q_[h]);
    }

    // ---- build A matrices: pair p packs head p (K=0..15) + head p+4 (K=16..31) ----
    // 16-bit A 16x32 layout (contiguous): v16h elem e -> K=16*hl+e
    half16 Amat[4];
#pragma unroll
    for (int p = 0; p < 4; ++p) {
        half16 v;
#pragma unroll
        for (int e = 0; e < 16; ++e) {
            const int h = hl ? (p + 4) : p;
            const int m = e;
            const int k = wave ? (m - row) : (row - m);   // decay exponent
            float val = 0.0f;
            if (k >= 0) val = a_[h] * exp2f((float)k * l2q_[h]);
            v[e] = (_Float16)val;
        }
        Amat[p] = v;
    }

    // per-lane head-group selected params (own head = p + 4*hl)
    float qsel[4], asel[4], q16sel[4], wsel[4];
    _Float16 wselh[4];
#pragma unroll
    for (int p = 0; p < 4; ++p) {
        qsel[p]   = hl ? q_[p + 4]   : q_[p];
        asel[p]   = hl ? a_[p + 4]   : a_[p];
        q16sel[p] = hl ? q16_[p + 4] : q16_[p];
        const int hg = (p + 4 * hl) + 8 * wave;
        const int dg = dt * DT + dl;
        const float w = E[(size_t)hg * DIMSZ + dg] * R[(size_t)hg * DIMSZ + dg];
        wsel[p]  = w;
        wselh[p] = (_Float16)w;
    }

    float chat[4] = {0.f, 0.f, 0.f, 0.f};   // weighted carries for own head group

#pragma unroll 1
    for (int ci = 0; ci < NCH; ++ci) {
        const int c = wave ? (NCH - 1 - ci) : ci;

        // B column: 16 timesteps of channel dl (same data in both lane halves)
        const half8 lo = *(const half8*)&xs[dl][c * CH];
        const half8 hi = *(const half8*)&xs[dl][c * CH + 8];
        half16 braw;
#pragma unroll
        for (int e = 0; e < 8; ++e) { braw[e] = lo[e]; braw[8 + e] = hi[e]; }

        // ---- 4 WMMAs: two heads each, W folded into B, fp32 accumulate ----
        float8 acc = {0.f, 0.f, 0.f, 0.f, 0.f, 0.f, 0.f, 0.f};
#pragma unroll
        for (int p = 0; p < 4; ++p) {
            half16 bw;
#pragma unroll
            for (int e = 0; e < 16; ++e) bw[e] = braw[e] * wselh[p];
            acc = __builtin_amdgcn_wmma_f32_16x16x32_f16(
                false, Amat[p], false, bw, (short)0, acc, false, false);
        }

        // ---- carry fixup: out[l,d] += sum_h q_h^(l+1) chat_h  (rev: q^(16-l)) ----
#pragma unroll
        for (int p = 0; p < 4; ++p) {
            const float cown = chat[p];
            const float coth = __shfl_xor(chat[p], 16, 32);
            const float qown = hl ? q_[p + 4] : q_[p];
            const float qoth = hl ? q_[p]     : q_[p + 4];
            float town, toth;
            if (!wave) {           // exponent (8*hl + 1) ascending in r
                town = hl ? q9_[p + 4] : q_[p];
                toth = hl ? q9_[p]     : q_[p + 4];
#pragma unroll
                for (int r = 0; r < 8; ++r) {
                    acc[r] = fmaf(town, cown, fmaf(toth, coth, acc[r]));
                    town *= qown; toth *= qoth;
                }
            } else {               // exponent (16 - l) -> start r=7 at q^(9-8*hl)
                town = hl ? q_[p + 4] : q9_[p];
                toth = hl ? q_[p]     : q9_[p + 4];
#pragma unroll
                for (int r = 7; r >= 0; --r) {
                    acc[r] = fmaf(town, cown, fmaf(toth, coth, acc[r]));
                    town *= qown; toth *= qoth;
                }
            }
        }

        // ---- carry update via Horner on the raw column ----
        float xf[16];
#pragma unroll
        for (int m = 0; m < 16; ++m) xf[m] = (float)braw[m];
        float H[4] = {0.f, 0.f, 0.f, 0.f};
        if (!wave) {
#pragma unroll
            for (int m = 0; m < 16; ++m) {
#pragma unroll
                for (int p = 0; p < 4; ++p) H[p] = fmaf(qsel[p], H[p], xf[m]);
            }
        } else {
#pragma unroll
            for (int m = 15; m >= 0; --m) {
#pragma unroll
                for (int p = 0; p < 4; ++p) H[p] = fmaf(qsel[p], H[p], xf[m]);
            }
        }
#pragma unroll
        for (int p = 0; p < 4; ++p)
            chat[p] = fmaf(q16sel[p], chat[p], wsel[p] * (asel[p] * H[p]));

        // ---- accumulate this direction's tile into LDS (2 deterministic addends) ----
#pragma unroll
        for (int r = 0; r < 8; ++r) {
            const int l = c * CH + 8 * hl + r;
            atomicAdd(&Obuf[l][dl], acc[r]);
        }
    }

    // export weighted segment carries
#pragma unroll
    for (int p = 0; p < 4; ++p) {
        const int h = p + 4 * hl;
        cws[CIDX(b, wave, s, h, dt * DT + dl)] = chat[p];
    }

    __syncthreads();
    // combined tile -> global, b128 stores
    float* ob = out + ((size_t)b * LSEQ + (size_t)s * SEG) * DIMSZ + (size_t)dt * DT;
    for (int i = threadIdx.x; i < SEG * DT / 4; i += 64) {
        const int t = i >> 2, c4 = (i & 3) * 4;
        float4 v;
        v.x = Obuf[t][c4 + 0];
        v.y = Obuf[t][c4 + 1];
        v.z = Obuf[t][c4 + 2];
        v.w = Obuf[t][c4 + 3];
        *(float4*)&ob[(size_t)t * DIMSZ + c4] = v;
    }
}

// ---------------------------------------------------------------------------
// K2: fold per-segment carries into per-segment carry-ins (length-8 serial scan).
// ---------------------------------------------------------------------------
__global__ __launch_bounds__(256) void k2_combine(
    const float* __restrict__ al,  const float* __restrict__ da,
    const float* __restrict__ ral, const float* __restrict__ rda,
    const float* __restrict__ cws, float* __restrict__ Pws)
{
    const int idx = blockIdx.x * blockDim.x + threadIdx.x;
    if (idx >= BSZ * 2 * HEADS_N * DIMSZ) return;
    const int d   = idx % DIMSZ;
    const int h   = (idx / DIMSZ) % HEADS_N;
    const int dir = (idx / (DIMSZ * HEADS_N)) % 2;
    const int b   = idx / (DIMSZ * HEADS_N * 2);

    const float av = sigm(dir ? ral[h] : al[h]);
    const float qv = (1.0f - av) * sigm(dir ? rda[h] : da[h]);
    const float Qs = __powf(qv, (float)SEG);

    float P = 0.0f;
    if (!dir) {
        for (int s2 = 0; s2 < NSEG; ++s2) {
            Pws[CIDX(b, 0, s2, h, d)] = P;
            P = cws[CIDX(b, 0, s2, h, d)] + Qs * P;
        }
    } else {
        for (int s2 = NSEG - 1; s2 >= 0; --s2) {
            Pws[CIDX(b, 1, s2, h, d)] = P;
            P = cws[CIDX(b, 1, s2, h, d)] + Qs * P;
        }
    }
}

// ---------------------------------------------------------------------------
// K3: out[b, s*SEG+t, d] += sum_h qf^(t+1)*Pf + qr^(SEG-t)*Pr   (exp2f: no
// running-product underflow poisoning).
// ---------------------------------------------------------------------------
__global__ __launch_bounds__(256) void k3_fixup(
    const float* __restrict__ al,  const float* __restrict__ da,
    const float* __restrict__ ral, const float* __restrict__ rda,
    const float* __restrict__ Pws, float* __restrict__ out)
{
    const int blk = blockIdx.x;
    const int dg  = blk % (DIMSZ / 256);
    const int s   = (blk / (DIMSZ / 256)) % NSEG;
    const int b   = blk / ((DIMSZ / 256) * NSEG);
    const int d   = dg * 256 + threadIdx.x;

    float Pf[HEADS_N], Pr[HEADS_N], lf[HEADS_N], lr[HEADS_N];
#pragma unroll
    for (int h = 0; h < HEADS_N; ++h) {
        const float af = sigm(al[h]);
        lf[h] = __log2f((1.0f - af) * sigm(da[h]));
        const float ar = sigm(ral[h]);
        lr[h] = __log2f((1.0f - ar) * sigm(rda[h]));
        Pf[h] = Pws[CIDX(b, 0, s, h, d)];
        Pr[h] = Pws[CIDX(b, 1, s, h, d)];
    }

    float* ob = out + ((size_t)b * LSEQ + (size_t)s * SEG) * DIMSZ;
#pragma unroll 1
    for (int t = 0; t < SEG; ++t) {
        float v = ob[(size_t)t * DIMSZ + d];
#pragma unroll
        for (int h = 0; h < HEADS_N; ++h) {
            v = fmaf(exp2f((float)(t + 1)   * lf[h]), Pf[h], v);
            v = fmaf(exp2f((float)(SEG - t) * lr[h]), Pr[h], v);
        }
        ob[(size_t)t * DIMSZ + d] = v;
    }
}

// ---------------------------------------------------------------------------
extern "C" void kernel_launch(void* const* d_in, const int* in_sizes, int n_in,
                              void* d_out, int out_size, void* d_ws, size_t ws_size,
                              hipStream_t stream)
{
    (void)in_sizes; (void)n_in; (void)out_size; (void)ws_size;
    const float* x   = (const float*)d_in[0];
    const float* E   = (const float*)d_in[1];
    const float* R   = (const float*)d_in[2];
    const float* al  = (const float*)d_in[3];
    const float* da  = (const float*)d_in[4];
    const float* ral = (const float*)d_in[5];
    const float* rda = (const float*)d_in[6];
    float* out = (float*)d_out;

    float* cws = (float*)d_ws;                                       // 1 MB
    float* Pws = cws + (size_t)BSZ * 2 * NSEG * HEADS_N * DIMSZ;     // 1 MB

    // K1: local chunked-scan (WMMA) — 1024 blocks x 64 threads
    k1_local<<<dim3(BSZ * NSEG * (DIMSZ / DT)), 64, 0, stream>>>(
        x, E, R, al, da, ral, rda, out, cws);

    // K2: carry combine across segments
    const int n2 = BSZ * 2 * HEADS_N * DIMSZ;
    k2_combine<<<dim3((n2 + 255) / 256), 256, 0, stream>>>(al, da, ral, rda, cws, Pws);

    // K3: cross-segment fixup (RMW of out)
    k3_fixup<<<dim3(BSZ * NSEG * (DIMSZ / 256)), 256, 0, stream>>>(
        al, da, ral, rda, Pws, out);
}
